// GCNBlock_29910152249793
// MI455X (gfx1250) — matrix operand order repaired and checked by reading, not compiled, not run on hardware
//
#include <hip/hip_runtime.h>

// ---------------------------------------------------------------------------
// GCN block for MI455X (gfx1250, wave32, WMMA).
// Dominant cost: 2 x (gather + scatter-add) edge aggregation -> HBM/L2 bound.
// GEMMs (4.9 GFLOP) run on v_wmma_f32_16x16x32_bf16; layer-1's
// bias+relu+dropout is fused into GEMM2's A-tile staging. A-streams use
// non-temporal loads so the L2 keeps the h matrix hot for the gather phase.
// ---------------------------------------------------------------------------

typedef __attribute__((ext_vector_type(16))) __bf16 v16bf;
typedef __attribute__((ext_vector_type(8)))  __bf16 v8bf;
typedef __attribute__((ext_vector_type(4)))  __bf16 v4bf;
typedef __attribute__((ext_vector_type(8)))  float  v8f;
typedef __attribute__((ext_vector_type(4)))  float  v4f;

#define HIDC 128
#define INC  256

__device__ __forceinline__ __bf16 f2bf(float f) {
  unsigned u = __builtin_bit_cast(unsigned, f);
  unsigned r = (u + 0x7FFFu + ((u >> 16) & 1u)) >> 16;   // RNE
  unsigned short s = (unsigned short)r;
  return __builtin_bit_cast(__bf16, s);
}

__device__ __forceinline__ float relu_drop(float v, unsigned idx, unsigned seed) {
  v = fmaxf(v, 0.0f);
  unsigned h = idx * 2654435761u ^ seed;
  h ^= h >> 16; h *= 0x7feb352du;
  h ^= h >> 15; h *= 0x846ca68bu;
  h ^= h >> 16;
  return (h >= 429496730u) ? v * (1.0f / 0.9f) : 0.0f;   // P(drop)=0.1
}

// ---------------- degree / norm --------------------------------------------
__global__ void k_deg_init(float* __restrict__ deg, int n) {
  int i = blockIdx.x * blockDim.x + threadIdx.x;
  if (i < n) deg[i] = 1.0f;                     // self-loop contribution
}

__global__ void k_deg_accum(const int* __restrict__ dst, float* __restrict__ deg, int E) {
  int i = blockIdx.x * blockDim.x + threadIdx.x;
  if (i < E) atomicAdd(&deg[dst[i]], 1.0f);
}

__global__ void k_deg_rsqrt(float* __restrict__ deg, int n) {
  int i = blockIdx.x * blockDim.x + threadIdx.x;
  if (i < n) deg[i] = rsqrtf(deg[i]);           // deg >= 1 always
}

// ---------------- weight convert: Wt[n][k] = bf16(W[k][n]) ------------------
__global__ void k_convert_wt(const float* __restrict__ W, __bf16* __restrict__ Wt,
                             int K, int Nc) {
  int i = blockIdx.x * blockDim.x + threadIdx.x;
  if (i < K * Nc) {
    int k = i / Nc, n = i % Nc;
    Wt[(size_t)n * K + k] = f2bf(W[(size_t)k * Nc + n]);
  }
}

// ---------------- WMMA GEMM: H[M x 128] = pre(A)[M x K] * W[K x 128] --------
// Block = 256 threads = 8 waves; TILES row-tiles of 16 per block; B fragments
// live in registers for the whole block; tile t+1's global loads are issued
// (into registers) while tile t's WMMA chain runs. Wave w owns cols [16w,16w+16).
// EPI=true applies bias+relu+dropout to A during staging (fused layer-1 epi).
template <int K, bool EPI>
__global__ __launch_bounds__(256)
void k_gemm_wmma(const float* __restrict__ A, const __bf16* __restrict__ Wt,
                 float* __restrict__ H, int M,
                 const float* __restrict__ bias, unsigned seed) {
  constexpr int KSTEPS = K / 32;
  constexpr int KQ     = K / 4;                 // float4s per row
  constexpr int SPT    = 16 * KQ / 256;         // staged float4s per thread
  constexpr int LDA    = K + 8;                 // padded bf16 row stride
  constexpr int TILES  = 4;
  __shared__ __attribute__((aligned(32))) __bf16 lA[16 * LDA];

  const int tid  = threadIdx.x;
  const int wave = tid >> 5;
  const int lane = tid & 31;

  // --- B fragments, registers for all K steps (one 32B v16bf load each) ----
  // B k order used here (contiguous, same as A): lanes 0-15 hold K=k0..k0+15 of col n, lanes 16-31 K+16.
  const int n = wave * 16 + (lane & 15);
  const int kBhalf = (lane >> 4) * 16;
  const __bf16* wrow = Wt + (size_t)n * K;
  v16bf Bf[KSTEPS];
#pragma unroll
  for (int ks = 0; ks < KSTEPS; ++ks)
    Bf[ks] = *reinterpret_cast<const v16bf*>(wrow + ks * 32 + kBhalf);

  const int arow = lane & 15;
  const int klo  = (lane >> 4) * 16;

  // --- software pipeline: tile t+1 global loads overlap tile t WMMAs -------
  v4f stage[SPT];
#pragma unroll
  for (int j = 0; j < SPT; ++j) {               // prologue: load tile 0
    int i = tid + j * 256;
    int rr = blockIdx.x * TILES * 16 + i / KQ;
    stage[j] = (rr < M)
        ? __builtin_nontemporal_load(
              reinterpret_cast<const v4f*>(A + (size_t)rr * K + (i % KQ) * 4))
        : (v4f){0.f, 0.f, 0.f, 0.f};
  }

  for (int t = 0; t < TILES; ++t) {
    const int row0 = (blockIdx.x * TILES + t) * 16;
    if (row0 >= M) break;                       // uniform across block
    if (t > 0) __syncthreads();                 // lA WAR vs previous tile

    // --- stage regs -> LDS (fp32 -> bf16, optional fused epilogue) ---------
#pragma unroll
    for (int j = 0; j < SPT; ++j) {
      int i  = tid + j * 256;
      int r  = i / KQ;
      int k4 = i % KQ;
      v4f f = stage[j];
      if constexpr (EPI) {
        int c = k4 * 4;                         // channel (K == HIDC here)
        unsigned base = (unsigned)(row0 + r) * (unsigned)K + (unsigned)c;
        f[0] = relu_drop(f[0] + bias[c + 0], base + 0, seed);
        f[1] = relu_drop(f[1] + bias[c + 1], base + 1, seed);
        f[2] = relu_drop(f[2] + bias[c + 2], base + 2, seed);
        f[3] = relu_drop(f[3] + bias[c + 3], base + 3, seed);
      }
      v4bf p = { f2bf(f[0]), f2bf(f[1]), f2bf(f[2]), f2bf(f[3]) };
      *reinterpret_cast<v4bf*>(&lA[r * LDA + k4 * 4]) = p;
    }
    __syncthreads();

    // --- issue next tile's global loads now (overlap with WMMA below) ------
    if (t + 1 < TILES && (blockIdx.x * TILES + t + 1) * 16 < M) {
#pragma unroll
      for (int j = 0; j < SPT; ++j) {
        int i = tid + j * 256;
        int rr = (blockIdx.x * TILES + t + 1) * 16 + i / KQ;
        stage[j] = (rr < M)
            ? __builtin_nontemporal_load(
                  reinterpret_cast<const v4f*>(A + (size_t)rr * K + (i % KQ) * 4))
            : (v4f){0.f, 0.f, 0.f, 0.f};
      }
    }

    // --- all A fragments first, then back-to-back WMMA chain ---------------
    // A k order used here (contiguous, same as B): lane l holds row (l&15); elems 0-7 = K klo..klo+7,
    // elems 8-15 = K klo+8..klo+15, klo = (l>>4)*16.
    v16bf Af[KSTEPS];
#pragma unroll
    for (int ks = 0; ks < KSTEPS; ++ks) {
      const __bf16* base = &lA[arow * LDA + ks * 32 + klo];
      v8bf lo = *reinterpret_cast<const v8bf*>(base);
      v8bf hi = *reinterpret_cast<const v8bf*>(base + 8);
#pragma unroll
      for (int i = 0; i < 8; ++i) { Af[ks][i] = lo[i]; Af[ks][i + 8] = hi[i]; }
    }
    v8f acc = {};
#pragma unroll
    for (int ks = 0; ks < KSTEPS; ++ks)
      acc = __builtin_amdgcn_wmma_f32_16x16x32_bf16(
          false, Af[ks], false, Bf[ks], (short)0, acc, false, false);

    // --- store C: VGPR j -> (M = row0 + j + 8*(lane>>4), N = lane&15) ------
    const int crow0 = row0 + (lane >> 4) * 8;
    const int ccol  = wave * 16 + (lane & 15);
    float* outp = H + (size_t)crow0 * HIDC + ccol;
    if (crow0 + 8 <= M) {                       // fast path: no per-row checks
#pragma unroll
      for (int j = 0; j < 8; ++j) outp[(size_t)j * HIDC] = acc[j];
    } else {
#pragma unroll
      for (int j = 0; j < 8; ++j)
        if (crow0 + j < M) outp[(size_t)j * HIDC] = acc[j];
    }
  }
}

// ---------------- edge aggregation (gather * norm -> scatter-add) -----------
// One wave per edge; edge/src/dst forced wave-uniform (SGPRs) so the row
// gather becomes scalar-base + lane-offset. Each lane moves 4 contiguous
// channels (512B coalesced row read, 4 f32 atomics into L2).
__global__ void k_aggregate(const float* __restrict__ h,
                            const int* __restrict__ src,
                            const int* __restrict__ dst,
                            const float* __restrict__ dinv,
                            float* __restrict__ agg, int E, int N) {
  int gid  = blockIdx.x * blockDim.x + threadIdx.x;
  int edge = gid >> 5;
  int c4   = (gid & 31) * 4;
  if (edge >= E + N) return;
  edge = __builtin_amdgcn_readfirstlane(edge);  // wave-uniform by construction
  int s, d;
  if (edge < E) {
    s = __builtin_amdgcn_readfirstlane(src[edge]);
    d = __builtin_amdgcn_readfirstlane(dst[edge]);
  } else {
    s = d = edge - E;                           // self-loop
  }
  float w = dinv[s] * dinv[d];
  const v4f v = *reinterpret_cast<const v4f*>(h + (size_t)s * HIDC + c4);
  float* out = agg + (size_t)d * HIDC + c4;
  atomicAdd(out + 0, v[0] * w);
  atomicAdd(out + 1, v[1] * w);
  atomicAdd(out + 2, v[2] * w);
  atomicAdd(out + 3, v[3] * w);
}

// ---------------- fused bias + relu + dropout (final layer, in place) -------
__global__ void k_bias_relu_dropout(float* __restrict__ io,
                                    const float* __restrict__ bias,
                                    int total, unsigned seed) {
  int i = blockIdx.x * blockDim.x + threadIdx.x;
  if (i >= total) return;
  float v = io[i] + bias[i & (HIDC - 1)];
  io[i] = relu_drop(v, (unsigned)i, seed);
}

// ---------------------------------------------------------------------------
static inline size_t align256(size_t x) { return (x + 255) & ~(size_t)255; }

extern "C" void kernel_launch(void* const* d_in, const int* in_sizes, int n_in,
                              void* d_out, int out_size, void* d_ws, size_t ws_size,
                              hipStream_t stream) {
  const float* x  = (const float*)d_in[0];
  const int*   ei = (const int*)d_in[1];
  const float* W1 = (const float*)d_in[2];
  const float* b1 = (const float*)d_in[3];
  const float* W2 = (const float*)d_in[4];
  const float* b2 = (const float*)d_in[5];

  const int N = in_sizes[0] / INC;              // 50000
  const int E = in_sizes[1] / 2;                // 800000
  const int* src = ei;
  const int* dst = ei + E;

  // workspace layout
  char*  ws   = (char*)d_ws;
  size_t szH  = align256((size_t)N * HIDC * sizeof(float));
  float* bufH   = (float*)(ws);                 // GEMM outputs (h1, then h2)
  float* bufAgg = (float*)(ws + szH);           // layer-1 raw aggregation
  float* dinv   = (float*)(ws + 2 * szH);       // degrees -> d^{-1/2}
  size_t szD    = align256((size_t)N * sizeof(float));
  __bf16* W1t   = (__bf16*)(ws + 2 * szH + szD);            // [128][256] bf16
  size_t szW1   = align256((size_t)HIDC * INC * sizeof(__bf16));
  __bf16* W2t   = (__bf16*)(ws + 2 * szH + szD + szW1);     // [128][128] bf16

  const int TB = 256;
  const int total = N * HIDC;
  const int aggThreads = (E + N) * 32;
  const int gemmBlocks = (N + 63) / 64;         // 4 row-tiles of 16 per block

  // 1) weights -> transposed bf16
  k_convert_wt<<<(INC * HIDC + TB - 1) / TB, TB, 0, stream>>>(W1, W1t, INC, HIDC);
  k_convert_wt<<<(HIDC * HIDC + TB - 1) / TB, TB, 0, stream>>>(W2, W2t, HIDC, HIDC);

  // 2) symmetric normalization
  k_deg_init <<<(N + TB - 1) / TB, TB, 0, stream>>>(dinv, N);
  k_deg_accum<<<(E + TB - 1) / TB, TB, 0, stream>>>(dst, dinv, E);
  k_deg_rsqrt<<<(N + TB - 1) / TB, TB, 0, stream>>>(dinv, N);

  // 3) layer 1: h1 = X @ W1  (WMMA bf16)
  k_gemm_wmma<INC, false><<<gemmBlocks, TB, 0, stream>>>(x, W1t, bufH, N,
                                                         nullptr, 0u);

  // 4) aggregate -> bufAgg (raw; epilogue deferred into GEMM2 staging)
  hipMemsetAsync(bufAgg, 0, (size_t)N * HIDC * sizeof(float), stream);
  k_aggregate<<<(aggThreads + TB - 1) / TB, TB, 0, stream>>>(bufH, src, dst, dinv,
                                                             bufAgg, E, N);

  // 5) layer 2: h2 = (bias+relu+dropout)(bufAgg) @ W2, epilogue fused in load
  k_gemm_wmma<HIDC, true><<<gemmBlocks, TB, 0, stream>>>(bufAgg, W2t, bufH, N,
                                                         b1, 0x9E3779B9u);

  // 6) aggregate directly into d_out
  hipMemsetAsync(d_out, 0, (size_t)N * HIDC * sizeof(float), stream);
  k_aggregate<<<(aggThreads + TB - 1) / TB, TB, 0, stream>>>(bufH, src, dst, dinv,
                                                             (float*)d_out, E, N);
  // 7) final bias + relu + dropout (in place on d_out)
  k_bias_relu_dropout<<<(total + TB - 1) / TB, TB, 0, stream>>>((float*)d_out, b2,
                                                                total, 0x85EBCA6Bu);
}
